// CondDecoder_6957847019937
// MI455X (gfx1250) — matrix operand order repaired and checked by reading, not compiled, not run on hardware
//
#include <hip/hip_runtime.h>
#include <hip/hip_fp16.h>

// ---------------------------------------------------------------------------
// CondDecoder (3-layer GRU + FC) for MI455X / gfx1250.
// wave32 WMMA f16 + TDM (tensor_load_to_lds) A-tile staging.
// Activation rows are ordered r = t*B + b so that one GEMM M-tile equals one
// GRU (batch-tile, timestep) tile; input-gate preactivations are kept in a
// blocked fragment-major layout [tile][lane][8] for coalesced store/reload.
// ---------------------------------------------------------------------------

typedef _Float16 h16;
typedef __attribute__((ext_vector_type(16))) _Float16     v16h;
typedef __attribute__((ext_vector_type(8)))  float        v8f;
typedef __attribute__((ext_vector_type(4)))  unsigned int u32x4;
typedef __attribute__((ext_vector_type(8)))  unsigned int u32x8;

#define B_   256
#define T_   128
#define V_   64
#define L_   128
#define H_   512
#define E_   32
#define IN0_ 161
#define K0P  192     // IN0 padded to multiple of 32
#define G3H  1536    // 3*H
#define M_   (B_*T_) // 32768 rows
#define NTG  96      // G3H/16 n-tiles in gi

// ---- WMMA fragment loaders (contiguous k = 16*half + s, same in A and B) ------------------
static __device__ __forceinline__ v16h load_frag2(const h16* p0, const h16* p1) {
  union { v16h v; uint4 q[2]; } f;
  f.q[0] = *(const uint4*)p0;
  f.q[1] = *(const uint4*)p1;
  return f.v;
}
static __device__ __forceinline__ v16h load_fragA(const h16* row, int k0, int half) {
  return load_frag2(row + k0 + half*16, row + k0 + half*16 + 8);
}
static __device__ __forceinline__ v16h load_fragB(const h16* row, int k0, int half) {
  const h16* p = row + k0 + half*16;
  return load_frag2(p, p + 8);
}
static __device__ __forceinline__ v8f wmma16(v16h a, v16h b, v8f c) {
  return __builtin_amdgcn_wmma_f32_16x16x32_f16(
      false, a, false, b, (short)0, c, false, false);
}
static __device__ __forceinline__ float sigmoidf_(float x) {
  return 1.0f / (1.0f + __expf(-x));
}

// ---- TDM: 1-D tile (16*K contiguous f16) global -> LDS offset 0 ------------
// D# group0: count=1, lds_addr=0, 57b global addr, type=2.
// D# group1: data_size=1 (2B), tensor_dim0=0x7FFFFFFF, tensor_dim1=1,
//            tile_dim0=16*K, tile_dim1/2 unused, dim0_stride=16*K.
// The LDS destination pointer is passed into the asm so the compiler knows
// the DMA writes it (otherwise loads from a never-stored LDS array fold away).
static __device__ __forceinline__ void tdm_load_atile(const h16* gsrc, unsigned elems,
                                                      h16* lds_dst) {
  unsigned long long ga = (unsigned long long)(const void*)gsrc;
  u32x4 g0 = { 1u, 0u, (unsigned)ga,
               ((unsigned)(ga >> 32) & 0x01FFFFFFu) | 0x80000000u };
  u32x8 g1 = { 0x00010000u,          // data_size=1 (2 bytes)
               0xFFFF0000u,          // tensor_dim0 lo16 << 16
               0x00017FFFu,          // tensor_dim0 hi16 | tensor_dim1(=1) lo16<<16
               (elems << 16),        // tensor_dim1 hi16(0) | tile_dim0<<16
               0u,                   // tile_dim1=0 (unused), tile_dim2=0
               elems,                // tensor_dim0_stride lo32
               0u, 0u };
  asm volatile("tensor_load_to_lds %0, %1"
               :
               : "s"(g0), "s"(g1), "v"(lds_dst)
               : "memory");
  __builtin_amdgcn_s_wait_tensorcnt(0);
}

// ---- helper kernels --------------------------------------------------------
__global__ void cvt_f32_to_f16(const float* __restrict__ in, h16* __restrict__ out, int n) {
  int i = blockIdx.x * blockDim.x + threadIdx.x;
  if (i < n) out[i] = (h16)in[i];
}

__global__ void pad_wih0(const float* __restrict__ W, h16* __restrict__ out) {
  int i = blockIdx.x * blockDim.x + threadIdx.x;
  if (i >= G3H * K0P) return;
  int n = i / K0P, k = i % K0P;
  out[i] = (k < IN0_) ? (h16)W[n * IN0_ + k] : (h16)0.0f;
}

// X0[t*B+b, 0:192] = [latent(128) | shifted_embed(32) | enthalpy(1) | 0...]
__global__ void build_x0(const float* __restrict__ latent, const float* __restrict__ enth,
                         const int* __restrict__ inp, const float* __restrict__ embed,
                         h16* __restrict__ X0) {
  int row = blockIdx.x;            // t*B + b
  int t = row / B_, b = row % B_;
  int k = threadIdx.x;             // 0..191
  float v;
  if (k < L_) {
    v = latent[b * L_ + k];
  } else if (k < L_ + E_) {
    int tok = (t == 0) ? 0 : inp[b * T_ + t - 1];
    v = embed[tok * E_ + (k - L_)];
  } else if (k == L_ + E_) {
    v = enth[b];
  } else {
    v = 0.0f;
  }
  X0[(size_t)row * K0P + k] = (h16)v;
}

// ---- input-gate GEMM with TDM-staged A tile --------------------------------
// Block = 24 waves (768 thr) covering all 1536 columns of one 16-row M-tile.
// A tile lives in DYNAMIC shared memory (16 KB, offset 0) and is DMA'd there
// exactly once per M-tile by the TDM.
__global__ void __launch_bounds__(768)
gemm_gi(const h16* __restrict__ A, const h16* __restrict__ W,
        const float* __restrict__ bias, float* __restrict__ gi_blk, int K) {
  extern __shared__ __align__(16) h16 atile[];   // dynamic LDS, offset 0
  int lane = threadIdx.x & 31;
  int wid  = threadIdx.x >> 5;          // 0..23 -> 64-column group
  int mt   = blockIdx.x;
  int n0   = wid * 64;
  int half = lane >> 4, nl = lane & 15;
  int mr   = mt << 4;

  if (wid == 0) tdm_load_atile(A + (size_t)mr * K, 16u * (unsigned)K, atile);
  __syncthreads();

  const h16* arow = &atile[nl * K];
  const h16* wr0  = W + (size_t)(n0      + nl) * K;
  const h16* wr1  = W + (size_t)(n0 + 16 + nl) * K;
  const h16* wr2  = W + (size_t)(n0 + 32 + nl) * K;
  const h16* wr3  = W + (size_t)(n0 + 48 + nl) * K;

  v8f acc0 = {}, acc1 = {}, acc2 = {}, acc3 = {};
  for (int k0 = 0; k0 < K; k0 += 32) {
    __builtin_prefetch(wr0 + k0 + 64, 0, 3);
    v16h a = load_fragA(arow, k0, half);       // ds_load_b128 x2
    acc0 = wmma16(a, load_fragB(wr0, k0, half), acc0);
    acc1 = wmma16(a, load_fragB(wr1, k0, half), acc1);
    acc2 = wmma16(a, load_fragB(wr2, k0, half), acc2);
    acc3 = wmma16(a, load_fragB(wr3, k0, half), acc3);
  }
  float b0 = bias[n0 + nl], b1 = bias[n0 + 16 + nl];
  float b2 = bias[n0 + 32 + nl], b3 = bias[n0 + 48 + nl];
  int nt0 = n0 >> 4;
  float* p0 = gi_blk + ((size_t)mt * NTG + nt0    ) * 256 + lane * 8;
  float* p1 = gi_blk + ((size_t)mt * NTG + nt0 + 1) * 256 + lane * 8;
  float* p2 = gi_blk + ((size_t)mt * NTG + nt0 + 2) * 256 + lane * 8;
  float* p3 = gi_blk + ((size_t)mt * NTG + nt0 + 3) * 256 + lane * 8;
  *(float4*)(p0)     = make_float4(acc0[0]+b0, acc0[1]+b0, acc0[2]+b0, acc0[3]+b0);
  *(float4*)(p0 + 4) = make_float4(acc0[4]+b0, acc0[5]+b0, acc0[6]+b0, acc0[7]+b0);
  *(float4*)(p1)     = make_float4(acc1[0]+b1, acc1[1]+b1, acc1[2]+b1, acc1[3]+b1);
  *(float4*)(p1 + 4) = make_float4(acc1[4]+b1, acc1[5]+b1, acc1[6]+b1, acc1[7]+b1);
  *(float4*)(p2)     = make_float4(acc2[0]+b2, acc2[1]+b2, acc2[2]+b2, acc2[3]+b2);
  *(float4*)(p2 + 4) = make_float4(acc2[4]+b2, acc2[5]+b2, acc2[6]+b2, acc2[7]+b2);
  *(float4*)(p3)     = make_float4(acc3[0]+b3, acc3[1]+b3, acc3[2]+b3, acc3[3]+b3);
  *(float4*)(p3 + 4) = make_float4(acc3[4]+b3, acc3[5]+b3, acc3[6]+b3, acc3[7]+b3);
}

// ---- persistent GRU recurrence --------------------------------------------
// One workgroup (32 waves) owns 16 batch rows for all T steps; h lives in LDS
// as f16 [16 x 512]. Wave w owns h columns [w*16, w*16+16): 3 gate tiles,
// K=512 -> 48 WMMAs/step. gi fragments are loaded (coalesced, blocked layout)
// before the K-loop so they overlap the matrix work. Y write-back is a
// cooperative coalesced LDS->global copy (one b128 per thread per step).
__global__ void __launch_bounds__(1024)
gru_layer(const float* __restrict__ gi_blk, const h16* __restrict__ Whh,
          const float* __restrict__ bhh, h16* __restrict__ Y) {
  __shared__ __align__(16) h16 hbuf[16 * H_];
  int lane = threadIdx.x & 31, w = threadIdx.x >> 5;
  int half = lane >> 4, nl = lane & 15;
  int g = blockIdx.x;                       // batch tile: b = g*16 + m

  for (int i = threadIdx.x; i < 16 * H_; i += 1024) hbuf[i] = (h16)0.0f;

  int col = (w << 4) + nl;
  float bh_r = bhh[col], bh_z = bhh[H_ + col], bh_n = bhh[2 * H_ + col];
  const h16* wr_r = Whh + (size_t)(col) * H_;
  const h16* wr_z = Whh + (size_t)(H_ + col) * H_;
  const h16* wr_n = Whh + (size_t)(2 * H_ + col) * H_;
  const h16* arow = &hbuf[nl * H_];

  // coalesced Y copy-out mapping: thread i -> (row i/64, 8 cols at (i%64)*8)
  int cr = threadIdx.x >> 6;
  int cc = (threadIdx.x & 63) * 8;
  const uint4* csrc = (const uint4*)&hbuf[cr * H_ + cc];

  float h[8];
#pragma unroll
  for (int v = 0; v < 8; ++v) h[v] = 0.0f;
  __syncthreads();

  for (int t = 0; t < T_; ++t) {
    // fragment-major gi loads for this (tile, step): 2x b128 per gate
    size_t mtrow = (size_t)(t * 16 + g) * NTG;
    const float* gpr = gi_blk + (mtrow +      w) * 256 + lane * 8;
    const float* gpz = gi_blk + (mtrow + 32 + w) * 256 + lane * 8;
    const float* gpn = gi_blk + (mtrow + 64 + w) * 256 + lane * 8;
    union { float4 f4[2]; float f[8]; } ir, iz, in_;
    ir.f4[0]  = *(const float4*)gpr;  ir.f4[1]  = *(const float4*)(gpr + 4);
    iz.f4[0]  = *(const float4*)gpz;  iz.f4[1]  = *(const float4*)(gpz + 4);
    in_.f4[0] = *(const float4*)gpn;  in_.f4[1] = *(const float4*)(gpn + 4);

    v8f ar = {}, az = {}, an = {};
#pragma unroll 4
    for (int k0 = 0; k0 < H_; k0 += 32) {
      v16h a = load_fragA(arow, k0, half);     // ds_load_b128 from shared h
      ar = wmma16(a, load_fragB(wr_r, k0, half), ar);
      az = wmma16(a, load_fragB(wr_z, k0, half), az);
      an = wmma16(a, load_fragB(wr_n, k0, half), an);
    }
    float hnew[8];
#pragma unroll
    for (int v = 0; v < 8; ++v) {
      float r = sigmoidf_(ir.f[v] + ar[v] + bh_r);
      float z = sigmoidf_(iz.f[v] + az[v] + bh_z);
      float n = tanhf(in_.f[v] + r * (an[v] + bh_n));
      hnew[v] = (1.0f - z) * n + z * h[v];
    }
    __syncthreads();                        // all waves done reading hbuf
#pragma unroll
    for (int v = 0; v < 8; ++v) {
      int rowm = (half << 3) + v;
      h[v] = hnew[v];
      hbuf[rowm * H_ + col] = (h16)hnew[v];
    }
    __syncthreads();                        // new h visible everywhere
    // coalesced write-back of this step's h tile: rows t*B + g*16 + cr
    *(uint4*)(Y + ((size_t)t * B_ + g * 16 + cr) * H_ + cc) = *csrc;
    // (read of hbuf; ordered before the next rewrite by the next barrier)
  }
}

// ---- FC: logits = Y * fcW^T + fcb, remapped to reference [b,t,v] order -----
__global__ void __launch_bounds__(128)
fc_gemm(const h16* __restrict__ A, const h16* __restrict__ W,
        const float* __restrict__ bias, float* __restrict__ out) {
  const int K = H_;
  int lane = threadIdx.x & 31;
  int mt   = blockIdx.x * 4 + (threadIdx.x >> 5);
  int half = lane >> 4, nl = lane & 15;
  int mr   = mt << 4;

  const h16* arow = A + (size_t)(mr + nl) * K;
  const h16* wr0  = W + (size_t)(     nl) * K;
  const h16* wr1  = W + (size_t)(16 + nl) * K;
  const h16* wr2  = W + (size_t)(32 + nl) * K;
  const h16* wr3  = W + (size_t)(48 + nl) * K;

  v8f acc0 = {}, acc1 = {}, acc2 = {}, acc3 = {};
  for (int k0 = 0; k0 < K; k0 += 32) {
    v16h a = load_fragA(arow, k0, half);
    acc0 = wmma16(a, load_fragB(wr0, k0, half), acc0);
    acc1 = wmma16(a, load_fragB(wr1, k0, half), acc1);
    acc2 = wmma16(a, load_fragB(wr2, k0, half), acc2);
    acc3 = wmma16(a, load_fragB(wr3, k0, half), acc3);
  }
  float b0 = bias[nl], b1 = bias[16 + nl], b2 = bias[32 + nl], b3 = bias[48 + nl];
#pragma unroll
  for (int v = 0; v < 8; ++v) {
    int r = mr + ((half << 3) + v);         // r = t*B + b
    int t = r >> 8, b = r & (B_ - 1);
    float* orow = out + ((size_t)b * T_ + t) * V_ + nl;
    orow[0]  = acc0[v] + b0;
    orow[16] = acc1[v] + b1;
    orow[32] = acc2[v] + b2;
    orow[48] = acc3[v] + b3;
  }
}

// ---------------------------------------------------------------------------

extern "C" void kernel_launch(void* const* d_in, const int* in_sizes, int n_in,
                              void* d_out, int out_size, void* d_ws, size_t ws_size,
                              hipStream_t stream) {
  (void)in_sizes; (void)n_in; (void)out_size; (void)ws_size;
  const float* latent = (const float*)d_in[0];
  const float* enth   = (const float*)d_in[1];
  const int*   inp    = (const int*)d_in[2];
  const float* embed  = (const float*)d_in[3];
  const float* Wih[3] = {(const float*)d_in[4],  (const float*)d_in[8],  (const float*)d_in[12]};
  const float* Whh[3] = {(const float*)d_in[5],  (const float*)d_in[9],  (const float*)d_in[13]};
  const float* bih[3] = {(const float*)d_in[6],  (const float*)d_in[10], (const float*)d_in[14]};
  const float* bhh[3] = {(const float*)d_in[7],  (const float*)d_in[11], (const float*)d_in[15]};
  const float* fcW    = (const float*)d_in[16];
  const float* fcb    = (const float*)d_in[17];

  char* ws = (char*)d_ws;
  size_t off = 0;
  auto carve = [&](size_t bytes) {
    void* p = ws + off;
    off += (bytes + 255) & ~(size_t)255;
    return p;
  };
  h16* Wih0p = (h16*)carve((size_t)G3H * K0P * sizeof(h16));
  h16* Wih1h = (h16*)carve((size_t)G3H * H_  * sizeof(h16));
  h16* Wih2h = (h16*)carve((size_t)G3H * H_  * sizeof(h16));
  h16* Whh0h = (h16*)carve((size_t)G3H * H_  * sizeof(h16));
  h16* Whh1h = (h16*)carve((size_t)G3H * H_  * sizeof(h16));
  h16* Whh2h = (h16*)carve((size_t)G3H * H_  * sizeof(h16));
  h16* fcWh  = (h16*)carve((size_t)V_  * H_  * sizeof(h16));
  h16* X0    = (h16*)carve((size_t)M_ * K0P * sizeof(h16));
  h16* Y1    = (h16*)carve((size_t)M_ * H_  * sizeof(h16));
  h16* Y2    = (h16*)carve((size_t)M_ * H_  * sizeof(h16));
  float* gi  = (float*)carve((size_t)M_ * G3H * sizeof(float));

  {
    int n = G3H * K0P;
    pad_wih0<<<(n + 255) / 256, 256, 0, stream>>>(Wih[0], Wih0p);
    n = G3H * H_;
    cvt_f32_to_f16<<<(n + 255) / 256, 256, 0, stream>>>(Wih[1], Wih1h, n);
    cvt_f32_to_f16<<<(n + 255) / 256, 256, 0, stream>>>(Wih[2], Wih2h, n);
    cvt_f32_to_f16<<<(n + 255) / 256, 256, 0, stream>>>(Whh[0], Whh0h, n);
    cvt_f32_to_f16<<<(n + 255) / 256, 256, 0, stream>>>(Whh[1], Whh1h, n);
    cvt_f32_to_f16<<<(n + 255) / 256, 256, 0, stream>>>(Whh[2], Whh2h, n);
    n = V_ * H_;
    cvt_f32_to_f16<<<(n + 255) / 256, 256, 0, stream>>>(fcW, fcWh, n);
  }

  build_x0<<<M_, K0P, 0, stream>>>(latent, enth, inp, embed, X0);

  const int MT = M_ >> 4;                       // 2048 M-tiles (one block each)
  const size_t atile_bytes = 16 * 512 * sizeof(h16);   // 16 KB dynamic LDS

  gemm_gi<<<MT, 768, atile_bytes, stream>>>(X0, Wih0p, bih[0], gi, K0P);
  gru_layer<<<B_ / 16, 1024, 0, stream>>>(gi, Whh0h, bhh[0], Y1);

  gemm_gi<<<MT, 768, atile_bytes, stream>>>(Y1, Wih1h, bih[1], gi, H_);
  gru_layer<<<B_ / 16, 1024, 0, stream>>>(gi, Whh1h, bhh[1], Y2);

  gemm_gi<<<MT, 768, atile_bytes, stream>>>(Y2, Wih2h, bih[2], gi, H_);
  gru_layer<<<B_ / 16, 1024, 0, stream>>>(gi, Whh2h, bhh[2], Y1);

  fc_gemm<<<MT / 4, 128, 0, stream>>>(Y1, fcWh, fcb, (float*)d_out);
}
